// MLPHead_77455440216266
// MI455X (gfx1250) — matrix operand order repaired and checked by reading, not compiled, not run on hardware
//
#include <hip/hip_runtime.h>
#include <cstddef>

// ---------------------------------------------------------------------------
// MLP-scored causal attention head, fused flash-style, bf16 WMMA on gfx1250.
// B=8, T=2048, C=1024, H=4096, HEAD=64.
// ---------------------------------------------------------------------------

#define B_    8
#define T_    2048
#define C_    1024
#define H_    4096
#define HEAD_ 64

typedef __bf16 bf16_t;
typedef bf16_t bf16x8 __attribute__((ext_vector_type(8)));
typedef bf16_t v16bf  __attribute__((ext_vector_type(16)));
typedef float  v8f    __attribute__((ext_vector_type(8)));

// ---- WMMA helpers ---------------------------------------------------------

__device__ __forceinline__ v8f wmma_bf16(v16bf a, v16bf b, v8f c) {
  // v_wmma_f32_16x16x32_bf16  D = A*B + C
  return __builtin_amdgcn_wmma_f32_16x16x32_bf16(
      /*neg_a=*/false, a, /*neg_b=*/false, b,
      /*c_mod=*/(short)0, c, /*reuse_a=*/false, /*reuse_b=*/false);
}

// A-matrix 16x32 bf16 fragment from row-major [M][K] storage (LDS or global).
// Lane half h: VGPR0-3 hold K = 8h+0..7, VGPR4-7 hold K = 8h+16..23.
__device__ __forceinline__ v16bf frag_a(const bf16_t* base, int ld, int m,
                                        int k0, int half) {
  const bf16_t* p = base + (size_t)m * ld + k0 + half * 8;
  bf16x8 c0 = *(const bf16x8*)(p);
  bf16x8 c1 = *(const bf16x8*)(p + 16);
  return __builtin_shufflevector(c0, c1, 0, 1, 2, 3, 4, 5, 6, 7,
                                 8, 9, 10, 11, 12, 13, 14, 15);
}

// B-matrix 32x16 bf16 fragment. Storage is transposed: row-major [N][K].
// Lane half h: VGPR0-3 hold K = 8h+0..7, VGPR4-7 hold K = 8h+16..23 for column n.
__device__ __forceinline__ v16bf frag_b(const bf16_t* base, int ld, int n,
                                        int k0, int half) {
  const bf16_t* p = base + (size_t)n * ld + k0 + half * 8;
  bf16x8 c0 = *(const bf16x8*)(p);
  bf16x8 c1 = *(const bf16x8*)(p + 16);
  return __builtin_shufflevector(c0, c1, 0, 1, 2, 3, 4, 5, 6, 7,
                                 8, 9, 10, 11, 12, 13, 14, 15);
}

// ---- Prep kernels: one-time fp32 -> bf16 (and transpose) into workspace ---

__global__ __launch_bounds__(256) void cvt_bf16_kernel(
    const float* __restrict__ in, bf16_t* __restrict__ out, int n) {
  int i = blockIdx.x * 256 + threadIdx.x;
  if (i < n) out[i] = (bf16_t)in[i];
}

__global__ __launch_bounds__(256) void transpose_cvt_kernel(
    const float* __restrict__ in, bf16_t* __restrict__ out, int rows, int cols) {
  int i = blockIdx.x * 256 + threadIdx.x;
  if (i < rows * cols) {
    int r = i / cols, c = i % cols;
    out[(size_t)c * rows + r] = (bf16_t)in[i];
  }
}

// ---- v^T = (x @ Wv)^T, bf16 ----------------------------------------------
// Output vt[b][d][t] so that p@v B-fragments are contiguous b128 loads.

__global__ __launch_bounds__(256) void value_kernel(
    const bf16_t* __restrict__ xb, const bf16_t* __restrict__ wvt,
    bf16_t* __restrict__ vt) {
  int tid = threadIdx.x;
  int w = tid >> 5, l = tid & 31, half = l >> 4, ln = l & 15;
  int task = blockIdx.x * 8 + w;      // (B*T/16) * (HEAD/16) = 4096 tasks
  int rowTile = task >> 2;
  int j = task & 3;
  int b = rowTile >> 7;
  int t0 = (rowTile & 127) << 4;
  int n0 = j * 16;
  const bf16_t* xrow = xb + (size_t)(b * T_ + t0) * C_;

  v8f acc = {0.f, 0.f, 0.f, 0.f, 0.f, 0.f, 0.f, 0.f};
  for (int k0 = 0; k0 < C_; k0 += 32) {
    v16bf a = frag_a(xrow, C_, ln, k0, half);
    v16bf bm = frag_b(wvt, C_, n0 + ln, k0, half);
    acc = wmma_bf16(a, bm, acc);
  }
  // D layout: lane covers col d = n0+ln, rows t0 + 8*half + r  (r = VGPR idx)
  int d = n0 + ln;
  bf16x8 pk;
#pragma unroll
  for (int r = 0; r < 8; ++r) pk[r] = (bf16_t)acc[r];
  *(bf16x8*)(vt + ((size_t)(b * HEAD_ + d)) * T_ + t0 + half * 8) = pk;
}

// ---- Fused: h = relu(x@W1+b1) in LDS; flash softmax(h@W2) @ v -------------

#define SMEM_H    (16 * H_ * 2)          /* 131072 */
#define SMEM_WEI  (8 * 16 * 32 * 4)      /*  16384 */
#define SMEM_P    (8 * 16 * 32 * 2)      /*   8192 */
#define SMEM_SC   (8 * 16 * 4)           /*    512 */
#define SMEM_TOTAL (SMEM_H + SMEM_WEI + SMEM_P + 3 * SMEM_SC) /* 157184 */

__global__ __launch_bounds__(256) void fused_attn_kernel(
    const bf16_t* __restrict__ xb, const bf16_t* __restrict__ w1t,
    const float* __restrict__ bias1, const bf16_t* __restrict__ w2t,
    const bf16_t* __restrict__ vt, float* __restrict__ out) {
  extern __shared__ char smem[];
  bf16_t* hB  = (bf16_t*)smem;                                  // [16][H_]
  float*  weiB = (float*)(smem + SMEM_H);                       // [8][16][32]
  bf16_t* pB   = (bf16_t*)(smem + SMEM_H + SMEM_WEI);           // [8][16][32]
  float*  scB  = (float*)(smem + SMEM_H + SMEM_WEI + SMEM_P);   // [8][16]
  float*  mF   = scB + 8 * 16;                                  // [8][16]
  float*  lF   = mF + 8 * 16;                                   // [8][16]
  float*  macc = (float*)smem;  // merge staging reuses dead h region [8][16][64]

  int tid = threadIdx.x;
  int w = tid >> 5, l = tid & 31, half = l >> 4, ln = l & 15;
  int blk = blockIdx.x;               // B*T/16 = 1024 blocks
  int b = blk >> 7;
  int t0 = (blk & 127) << 4;
  const bf16_t* xrow = xb + (size_t)(b * T_ + t0) * C_;

  // ---------------- Phase 1: h tile [16][4096] -> LDS (bf16) --------------
  // Wave w owns H columns [512w, 512w+512): 32 WMMA n-tiles, K=1024.
  for (int i = 0; i < 32; ++i) {
    int n0 = (w * 32 + i) * 16;
    v8f acc = {0.f, 0.f, 0.f, 0.f, 0.f, 0.f, 0.f, 0.f};
    for (int k0 = 0; k0 < C_; k0 += 32) {
      v16bf a = frag_a(xrow, C_, ln, k0, half);
      v16bf bm = frag_b(w1t, C_, n0 + ln, k0, half);
      acc = wmma_bf16(a, bm, acc);
    }
    float bv = bias1[n0 + ln];
#pragma unroll
    for (int r = 0; r < 8; ++r) {
      float hv = fmaxf(acc[r] + bv, 0.f);
      hB[(r + half * 8) * H_ + n0 + ln] = (bf16_t)hv;
    }
  }
  __syncthreads();

  // ---------------- Phase 2: flash loop over 32-wide causal col blocks ----
  v8f o0 = {0.f,0.f,0.f,0.f,0.f,0.f,0.f,0.f};
  v8f o1 = o0, o2 = o0, o3 = o0;
  float mprev = -INFINITY, lprev = 0.f;
  float*  weiW = weiB + w * 16 * 32;
  bf16_t* pW   = pB + w * 16 * 32;
  float*  scW  = scB + w * 16;
  const bf16_t* vbase = vt + (size_t)b * HEAD_ * T_;
  int sbMax = (t0 + 15) >> 5;

  for (int sb = w; sb <= sbMax; sb += 8) {
    int s0 = sb << 5;
    // wei tile [16 rows][32 cols] = h @ W2[:, s0:s0+32]
    v8f wa0 = {0.f,0.f,0.f,0.f,0.f,0.f,0.f,0.f};
    v8f wa1 = wa0;
    for (int k0 = 0; k0 < H_; k0 += 32) {
      v16bf a  = frag_a(hB, H_, ln, k0, half);
      v16bf bf0 = frag_b(w2t, H_, s0 + ln, k0, half);
      v16bf bf1 = frag_b(w2t, H_, s0 + 16 + ln, k0, half);
      wa0 = wmma_bf16(a, bf0, wa0);
      wa1 = wmma_bf16(a, bf1, wa1);
    }
    // stage wei tile to LDS (D-layout scatter)
#pragma unroll
    for (int r = 0; r < 8; ++r) {
      int row = r + half * 8;
      weiW[row * 32 + ln] = wa0[r];
      weiW[row * 32 + 16 + ln] = wa1[r];
    }
    asm volatile("s_wait_dscnt 0" ::: "memory");

    // Row pass: lane owns (row ln, cols 16*half..16*half+15)
    int trow = t0 + ln;
    float wv[16];
    float tmax = -INFINITY;
#pragma unroll
    for (int i2 = 0; i2 < 16; ++i2) {
      int cg = s0 + half * 16 + i2;
      float xv = weiW[ln * 32 + half * 16 + i2];
      xv = (cg <= trow) ? xv : -INFINITY;
      wv[i2] = xv;
      tmax = fmaxf(tmax, xv);
    }
    tmax = fmaxf(tmax, __shfl_xor(tmax, 16, 32));
    float mn = fmaxf(mprev, tmax);
    float msafe = (mn == -INFINITY) ? 0.f : mn;
    float ps = 0.f;
#pragma unroll
    for (int i2 = 0; i2 < 16; ++i2) {
      float p = __expf(wv[i2] - msafe);
      ps += p;
      pW[ln * 32 + half * 16 + i2] = (bf16_t)p;
    }
    ps += __shfl_xor(ps, 16, 32);
    float scale = __expf(mprev - msafe);
    lprev = lprev * scale + ps;
    mprev = mn;
    scW[ln] = scale;                      // halves write identical values
    asm volatile("s_wait_dscnt 0" ::: "memory");

    // rescale running output accumulators (per-row scale, D layout)
    float sc[8];
#pragma unroll
    for (int r = 0; r < 8; ++r) sc[r] = scW[r + half * 8];
#pragma unroll
    for (int r = 0; r < 8; ++r) {
      o0[r] *= sc[r]; o1[r] *= sc[r]; o2[r] *= sc[r]; o3[r] *= sc[r];
    }

    // out += p[16x32] @ v[32x64]  (one WMMA per 16-wide head tile)
    v16bf pa = frag_a(pW, 32, ln, 0, half);
    v16bf vb0 = frag_b(vbase, T_, 0 + ln, s0, half);
    v16bf vb1 = frag_b(vbase, T_, 16 + ln, s0, half);
    v16bf vb2 = frag_b(vbase, T_, 32 + ln, s0, half);
    v16bf vb3 = frag_b(vbase, T_, 48 + ln, s0, half);
    o0 = wmma_bf16(pa, vb0, o0);
    o1 = wmma_bf16(pa, vb1, o1);
    o2 = wmma_bf16(pa, vb2, o2);
    o3 = wmma_bf16(pa, vb3, o3);
  }

  __syncthreads();  // all phase-2 reads of hB done -> safe to reuse as macc

  // stage per-wave partials
#pragma unroll
  for (int r = 0; r < 8; ++r) {
    int row = r + half * 8;
    float* dst = macc + (w * 16 + row) * 64;
    dst[ln] = o0[r];
    dst[16 + ln] = o1[r];
    dst[32 + ln] = o2[r];
    dst[48 + ln] = o3[r];
  }
  if (l < 16) {
    mF[w * 16 + ln] = mprev;
    lF[w * 16 + ln] = lprev;
  }
  __syncthreads();

  // merge 8 wave partials: thread -> (row = tid>>4, 4 head cols)
  {
    int r = tid >> 4;
    int c4 = (tid & 15) << 2;
    float M = -INFINITY;
#pragma unroll
    for (int ww = 0; ww < 8; ++ww) M = fmaxf(M, mF[ww * 16 + r]);
    float L = 0.f;
    float o[4] = {0.f, 0.f, 0.f, 0.f};
#pragma unroll
    for (int ww = 0; ww < 8; ++ww) {
      float s = __expf(mF[ww * 16 + r] - M);
      L += s * lF[ww * 16 + r];
      const float* src = macc + (ww * 16 + r) * 64 + c4;
      o[0] += s * src[0];
      o[1] += s * src[1];
      o[2] += s * src[2];
      o[3] += s * src[3];
    }
    float inv = 1.f / L;
    float4 res = make_float4(o[0] * inv, o[1] * inv, o[2] * inv, o[3] * inv);
    *(float4*)(out + (size_t)(b * T_ + t0 + r) * HEAD_ + c4) = res;
  }
}

// ---------------------------------------------------------------------------

extern "C" void kernel_launch(void* const* d_in, const int* in_sizes, int n_in,
                              void* d_out, int out_size, void* d_ws, size_t ws_size,
                              hipStream_t stream) {
  (void)in_sizes; (void)n_in; (void)out_size; (void)ws_size;
  const float* x  = (const float*)d_in[0];   // [B,T,C]
  const float* W1 = (const float*)d_in[1];   // [C,H]
  const float* b1 = (const float*)d_in[2];   // [H]
  const float* W2 = (const float*)d_in[3];   // [H,T]
  const float* Wv = (const float*)d_in[4];   // [C,HEAD]
  float* out = (float*)d_out;                // [B,T,HEAD] f32

  // Workspace layout (bf16), total ~58 MB
  char* ws = (char*)d_ws;
  const size_t xb_bytes  = (size_t)B_ * T_ * C_ * 2;    // 33,554,432
  const size_t w1t_bytes = (size_t)H_ * C_ * 2;         //  8,388,608
  const size_t w2t_bytes = (size_t)T_ * H_ * 2;         // 16,777,216
  const size_t wvt_bytes = (size_t)HEAD_ * C_ * 2;      //    131,072
  bf16_t* xb  = (bf16_t*)ws;
  bf16_t* w1t = (bf16_t*)(ws + xb_bytes);
  bf16_t* w2t = (bf16_t*)(ws + xb_bytes + w1t_bytes);
  bf16_t* wvt = (bf16_t*)(ws + xb_bytes + w1t_bytes + w2t_bytes);
  bf16_t* vt  = (bf16_t*)(ws + xb_bytes + w1t_bytes + w2t_bytes + wvt_bytes);

  const int nx = B_ * T_ * C_;    // 16,777,216
  cvt_bf16_kernel<<<(nx + 255) / 256, 256, 0, stream>>>(x, xb, nx);
  transpose_cvt_kernel<<<(C_ * H_ + 255) / 256, 256, 0, stream>>>(W1, w1t, C_, H_);
  transpose_cvt_kernel<<<(H_ * T_ + 255) / 256, 256, 0, stream>>>(W2, w2t, H_, T_);
  transpose_cvt_kernel<<<(C_ * HEAD_ + 255) / 256, 256, 0, stream>>>(Wv, wvt, C_, HEAD_);

  value_kernel<<<(B_ * T_ / 16) * (HEAD_ / 16) / 8, 256, 0, stream>>>(xb, wvt, vt);

  fused_attn_kernel<<<B_ * T_ / 16, 256, SMEM_TOTAL, stream>>>(
      xb, w1t, b1, w2t, vt, out);
}
